// ScaledDotProductAttention_36481452212425
// MI455X (gfx1250) — matrix operand order repaired and checked by reading, not compiled, not run on hardware
//
#include <hip/hip_runtime.h>

// ---- problem constants (from reference): B=2, H=16, S=2048, D=64, causal ----
constexpr int S_LEN  = 2048;
constexpr int HEAD_D = 64;
constexpr int BH     = 2 * 16;
constexpr int Q_BLOCK = 128;   // 8 waves * 16 rows
constexpr int KV_STEP = 32;

// softmax in base 2: scale = (1/sqrt(64)) * log2(e)
#define SCL 0.18033688011112042f

typedef __bf16 bf16_t;
typedef bf16_t v16bf __attribute__((ext_vector_type(16)));
typedef float  v8f   __attribute__((ext_vector_type(8)));
typedef unsigned int v4u __attribute__((ext_vector_type(4)));
typedef int v8i __attribute__((ext_vector_type(8)));
typedef int v4i __attribute__((ext_vector_type(4)));

#if __has_builtin(__builtin_amdgcn_tensor_load_to_lds)
#define HAVE_TDM 1
#else
#define HAVE_TDM 0
#endif

union FragB {
    v16bf v;
    unsigned int u[8];
};

// ---------- helpers ----------
__device__ __forceinline__ float fast_exp2(float x) {
#if __has_builtin(__builtin_amdgcn_exp2f)
    return __builtin_amdgcn_exp2f(x);
#else
    return exp2f(x);
#endif
}

__device__ __forceinline__ unsigned short f2bf(float x) {
    unsigned int u = __builtin_bit_cast(unsigned int, x);
    u += 0x7FFFu + ((u >> 16) & 1u);
    return (unsigned short)(u >> 16);
}

__device__ __forceinline__ unsigned int packbf2(float lo, float hi) {
#if __has_builtin(__builtin_amdgcn_cvt_pk_bf16_f32)
    auto p = __builtin_amdgcn_cvt_pk_bf16_f32(lo, hi);
    return __builtin_bit_cast(unsigned int, p);
#else
    return (unsigned int)f2bf(lo) | ((unsigned int)f2bf(hi) << 16);
#endif
}

// max-reduce across each 16-lane row (result replicated to all 16 lanes)
#if __has_builtin(__builtin_amdgcn_permlane16)
__device__ __forceinline__ float pl16_max(float x, int s0, int s1) {
    int xi = __builtin_bit_cast(int, x);
    int yi = __builtin_amdgcn_permlane16(xi, xi, s0, s1, false, false);
    return fmaxf(x, __builtin_bit_cast(float, yi));
}
__device__ __forceinline__ float redmax16(float x) {
    x = pl16_max(x, 0x67452301, 0xEFCDAB89);  // xor 1
    x = pl16_max(x, 0x54761032, 0xDCFE98BA);  // xor 2
    x = pl16_max(x, 0x32107654, 0xBA98FEDC);  // xor 4
    x = pl16_max(x, 0xFEDCBA98, 0x76543210);  // xor 8
    return x;
}
#else
__device__ __forceinline__ float redmax16(float x) {
#pragma unroll
    for (int off = 1; off < 16; off <<= 1)
        x = fmaxf(x, __shfl_xor(x, off, 16));
    return x;
}
#endif

__device__ __forceinline__ v8f wmma_bf16(const FragB& a, const FragB& b, v8f c) {
    return __builtin_amdgcn_wmma_f32_16x16x32_bf16(
        false, a.v, false, b.v, (short)0, c, false, false);
}

#if HAVE_TDM
// Issue one 2D TDM tile load (bf16 elements) into LDS. Dims in 2-byte units.
// This toolchain exposes the 6-arg builtin: (v4u, v8i, v4i, v4i, v8i, i32 cpol).
__device__ __forceinline__ void tdm_load_2d(unsigned lds_addr, unsigned long long gaddr,
                                            unsigned tensor_d0, unsigned tensor_d1,
                                            unsigned tile_d0, unsigned tile_d1,
                                            unsigned long long stride0) {
    v4u g0;
    g0[0] = 1u;                                             // count=1, user D#
    g0[1] = lds_addr;                                       // LDS byte address
    g0[2] = (unsigned)(gaddr & 0xFFFFFFFFu);                // global_addr[31:0]
    g0[3] = (unsigned)((gaddr >> 32) & 0x1FFFFFFu) | (2u << 30);  // addr[56:32], type=2
    v8i g1;
    g1[0] = (int)(1u << 16);                                // wg_mask=0, data_size=2B
    g1[1] = (int)((tensor_d0 & 0xFFFFu) << 16);             // tensor_dim0 lo
    g1[2] = (int)((tensor_d0 >> 16) | ((tensor_d1 & 0xFFFFu) << 16));
    g1[3] = (int)((tensor_d1 >> 16) | (tile_d0 << 16));     // tile_dim0
    g1[4] = (int)(tile_d1 & 0xFFFFu);                       // tile_dim1 (tile_dim2=0)
    g1[5] = (int)(unsigned)(stride0 & 0xFFFFFFFFu);         // tensor_dim0_stride lo
    g1[6] = (int)(unsigned)((stride0 >> 32) & 0xFFFFu);     // stride0 hi (stride1=0)
    g1[7] = 0;
    v4i g2 = {0, 0, 0, 0};
    v4i g3 = {0, 0, 0, 0};
    v8i g4 = {0, 0, 0, 0, 0, 0, 0, 0};
    __builtin_amdgcn_tensor_load_to_lds(g0, g1, g2, g3, g4, 0);
}
__device__ __forceinline__ void tdm_wait0() {
#if __has_builtin(__builtin_amdgcn_s_wait_tensorcnt)
    __builtin_amdgcn_s_wait_tensorcnt(0);
#else
    asm volatile("s_wait_tensorcnt 0x0" ::: "memory");
#endif
}
#endif

// ================= pre-pass: fp32 -> bf16, V transposed =================
// ws layout (bf16): Qbf[BH][S][D] | Kbf[BH][S][D] | Vt[BH][D][S]
__global__ __launch_bounds__(256)
void convert_kernel(const float* __restrict__ Q, const float* __restrict__ K,
                    const float* __restrict__ V, unsigned short* __restrict__ ws) {
    const size_t seg = (size_t)BH * S_LEN * HEAD_D;         // elements per tensor
    const int idx4 = blockIdx.x * 256 + threadIdx.x;        // one float4 per thread
    const size_t flat = (size_t)idx4 * 4;

    unsigned int* Qw = reinterpret_cast<unsigned int*>(ws);
    unsigned int* Kw = reinterpret_cast<unsigned int*>(ws + seg);
    unsigned short* Vt = ws + 2 * seg;

    float4 q = reinterpret_cast<const float4*>(Q)[idx4];
    float4 k = reinterpret_cast<const float4*>(K)[idx4];
    float4 v = reinterpret_cast<const float4*>(V)[idx4];

    Qw[idx4 * 2 + 0] = packbf2(q.x, q.y);
    Qw[idx4 * 2 + 1] = packbf2(q.z, q.w);
    Kw[idx4 * 2 + 0] = packbf2(k.x, k.y);
    Kw[idx4 * 2 + 1] = packbf2(k.z, k.w);

    // V transpose: flat = ((bh*S + s)*64 + d), d multiple of 4
    const int bh = (int)(flat >> 17);          // S*D = 131072 = 2^17
    const int rem = (int)(flat & 0x1FFFF);
    const int s  = rem >> 6;
    const int d  = rem & 63;
    unsigned int p01 = packbf2(v.x, v.y);
    unsigned int p23 = packbf2(v.z, v.w);
    unsigned short* vb = Vt + ((size_t)bh * HEAD_D + d) * S_LEN + s;
    vb[0 * S_LEN] = (unsigned short)(p01 & 0xFFFFu);
    vb[1 * S_LEN] = (unsigned short)(p01 >> 16);
    vb[2 * S_LEN] = (unsigned short)(p23 & 0xFFFFu);
    vb[3 * S_LEN] = (unsigned short)(p23 >> 16);
}

// ================= flash attention (bf16 WMMA, causal) =================
__global__ __launch_bounds__(256)
void fa_causal_kernel(const unsigned short* __restrict__ ws, float* __restrict__ O) {
    __shared__ unsigned int sK[2][32 * 32];  // [buf][key][d-pair] bf16x2
    __shared__ unsigned int sV[2][64 * 16];  // [buf][d][key-pair] bf16x2 (V^T tile)
    __shared__ unsigned int sP[8 * 256];     // per-wave 16x32 bf16

    const size_t seg = (size_t)BH * S_LEN * HEAD_D;
    const int tid  = threadIdx.x;
    const int lane = tid & 31;
    const int wave = tid >> 5;
    const int n    = lane & 15;
    const int h    = lane >> 4;

    const int qtiles = S_LEN / Q_BLOCK;
    const int bh = blockIdx.x / qtiles;
    const int qb = (blockIdx.x % qtiles) * Q_BLOCK;
    const int qw = qb + wave * 16;
    const int qwS = __builtin_amdgcn_readfirstlane(qw);  // wave-uniform for scalar branches

    const unsigned int*   Qw = reinterpret_cast<const unsigned int*>(ws) + (size_t)bh * S_LEN * 32;
    const unsigned short* Kb = ws + seg + (size_t)bh * S_LEN * HEAD_D;
    const unsigned short* Vt = ws + 2 * seg + (size_t)bh * (size_t)HEAD_D * S_LEN;
    float* Ob = O + (size_t)bh * S_LEN * HEAD_D;

    // ---- Q fragments: raw bf16 dwords, no conversion ----
    FragB aQ[2];
    {
        const unsigned int* qp = Qw + (size_t)(qw + n) * 32;
#pragma unroll
        for (int ch = 0; ch < 2; ++ch)
#pragma unroll
            for (int v = 0; v < 8; ++v) {
                const int kb2 = v + 8 * h;
                aQ[ch].u[v] = qp[ch * 16 + kb2];
            }
    }

    float mrow[8], lrow[8], alpha[8];
    v8f acc[4];
#pragma unroll
    for (int r = 0; r < 8; ++r) { mrow[r] = -__builtin_inff(); lrow[r] = 0.0f; }
#pragma unroll
    for (int d = 0; d < 4; ++d)
#pragma unroll
        for (int r = 0; r < 8; ++r) acc[d][r] = 0.0f;

    FragB bones;
#pragma unroll
    for (int v = 0; v < 8; ++v) bones.u[v] = 0x3F803F80u;  // bf16 1.0 x2

    // ---- staging: TDM (wave 0 drives both tiles) or cooperative copy ----
    auto issueStage = [&](int tile, int buf) {
#if HAVE_TDM
        if (wave == 0) {
            // K tile: 32 keys x 64 d, fully contiguous
            tdm_load_2d((unsigned)(size_t)&sK[buf][0],
                        (unsigned long long)(size_t)(Kb + (size_t)tile * KV_STEP * HEAD_D),
                        /*tensor_d0=*/HEAD_D, /*tensor_d1=*/KV_STEP,
                        /*tile_d0=*/HEAD_D,   /*tile_d1=*/KV_STEP,
                        /*stride0=*/HEAD_D);
            // V^T tile: 64 d-rows x 32 keys, row stride S
            tdm_load_2d((unsigned)(size_t)&sV[buf][0],
                        (unsigned long long)(size_t)(Vt + (size_t)tile * KV_STEP),
                        /*tensor_d0=*/KV_STEP, /*tensor_d1=*/HEAD_D,
                        /*tile_d0=*/KV_STEP,   /*tile_d1=*/HEAD_D,
                        /*stride0=*/S_LEN);
        }
#else
        // cooperative byte copy: 4KB each, one uint4 per thread
        uint4 kq = reinterpret_cast<const uint4*>(Kb + (size_t)tile * KV_STEP * HEAD_D)[tid];
        const int d = tid >> 2, c = tid & 3;
        uint4 vq = reinterpret_cast<const uint4*>(Vt + (size_t)d * S_LEN + (size_t)tile * KV_STEP)[c];
        reinterpret_cast<uint4*>(&sK[buf][0])[tid] = kq;
        reinterpret_cast<uint4*>(&sV[buf][0])[tid] = vq;
#endif
    };
    auto finishStage = [&]() {
#if HAVE_TDM
        if (wave == 0) tdm_wait0();
#endif
    };

    auto computeTile = [&](int kv, int buf) {
        FragB bk0l, bk0h, bk1l, bk1h;
#pragma unroll
        for (int v = 0; v < 8; ++v) {
            const unsigned int* kbuf = sK[buf];
            bk0l.u[v] = kbuf[(n)      * 32 +      8 * h + v];
            bk0h.u[v] = kbuf[(n)      * 32 + 16 + 8 * h + v];
            bk1l.u[v] = kbuf[(16 + n) * 32 +      8 * h + v];
            bk1h.u[v] = kbuf[(16 + n) * 32 + 16 + 8 * h + v];
        }
        v8f z;
#pragma unroll
        for (int r = 0; r < 8; ++r) z[r] = 0.0f;
        v8f s0 = wmma_bf16(aQ[0], bk0l, wmma_bf16(aQ[1], bk0h, z));
        v8f s1 = wmma_bf16(aQ[0], bk1l, wmma_bf16(aQ[1], bk1h, z));

        const bool needmask = (kv + 31 > qwS);  // scalar: tile touches diagonal
        unsigned short* Pw16 = reinterpret_cast<unsigned short*>(sP + wave * 256);
#pragma unroll
        for (int r = 0; r < 8; ++r) {
            float s0r = s0[r] * SCL;
            float s1r = s1[r] * SCL;
            if (needmask) {
                const int row = qw + r + 8 * h;
                s0r = (kv + n      > row) ? -__builtin_inff() : s0r;
                s1r = (kv + 16 + n > row) ? -__builtin_inff() : s1r;
            }
            const float t  = redmax16(fmaxf(s0r, s1r));
            const float mo = mrow[r];
            const float mn = fmaxf(mo, t);
            alpha[r] = fast_exp2(mo - mn);
            mrow[r]  = mn;
            const float p0 = fast_exp2(s0r - mn);
            const float p1 = fast_exp2(s1r - mn);
            unsigned int pk = packbf2(p0, p1);
            Pw16[(r + 8 * h) * 32 +      n] = (unsigned short)(pk & 0xFFFFu);
            Pw16[(r + 8 * h) * 32 + 16 + n] = (unsigned short)(pk >> 16);
        }

        asm volatile("s_wait_dscnt 0x0" ::: "memory");  // wave-private LDS RAW

        FragB aP;
        const unsigned int* Pw32 = sP + wave * 256;
#pragma unroll
        for (int v = 0; v < 8; ++v) {
            const int kb2 = v + 8 * h;
            aP.u[v] = Pw32[n * 16 + kb2];
        }

        v8f rs = wmma_bf16(aP, bones, z);               // row sums on the matrix pipe
#pragma unroll
        for (int r = 0; r < 8; ++r)
            lrow[r] = lrow[r] * alpha[r] + rs[r];

#pragma unroll
        for (int dt = 0; dt < 4; ++dt) {
            FragB bv;
#pragma unroll
            for (int v = 0; v < 8; ++v)
                bv.u[v] = sV[buf][(dt * 16 + n) * 16 + 8 * h + v];
#pragma unroll
            for (int r = 0; r < 8; ++r)
                acc[dt][r] *= alpha[r];
            acc[dt] = wmma_bf16(aP, bv, acc[dt]);
        }
    };

    // ---- software-pipelined main loop, 1 barrier / tile ----
    const int nT = (qb + Q_BLOCK) / KV_STEP;
    issueStage(0, 0);
    finishStage();
    __syncthreads();

    for (int it = 0; it < nT; ++it) {
        const int kv   = it * KV_STEP;
        const bool more = (it + 1) < nT;
        if (more) issueStage(it + 1, (it + 1) & 1);   // copy next tile while computing
        if (kv <= qwS + 15) computeTile(kv, it & 1);  // per-wave causal skip (scalar)
        if (more) finishStage();
        __syncthreads();
    }

    // ---- normalize and store ----
    float inv[8];
#pragma unroll
    for (int r = 0; r < 8; ++r) inv[r] = 1.0f / lrow[r];
#pragma unroll
    for (int dt = 0; dt < 4; ++dt)
#pragma unroll
        for (int r = 0; r < 8; ++r) {
            const int row = qw + r + 8 * h;
            Ob[(size_t)row * HEAD_D + dt * 16 + n] = acc[dt][r] * inv[r];
        }
}

extern "C" void kernel_launch(void* const* d_in, const int* in_sizes, int n_in,
                              void* d_out, int out_size, void* d_ws, size_t ws_size,
                              hipStream_t stream) {
    const float* Q = (const float*)d_in[0];
    const float* K = (const float*)d_in[1];
    const float* V = (const float*)d_in[2];
    float* O = (float*)d_out;
    unsigned short* ws = (unsigned short*)d_ws;

    // pre-pass: fp32 -> bf16 (Q,K) + transposed bf16 V, into d_ws (~24 MB)
    const int convBlocks = (BH * S_LEN * HEAD_D / 4) / 256;  // 4096
    convert_kernel<<<dim3(convBlocks), dim3(256), 0, stream>>>(Q, K, V, ws);

    const int blocks = BH * (S_LEN / Q_BLOCK);  // 512
    fa_causal_kernel<<<dim3(blocks), dim3(256), 0, stream>>>(ws, O);
}
